// RSSMWorldModel_69423851372556
// MI455X (gfx1250) — matrix operand order repaired and checked by reading, not compiled, not run on hardware
//
#include <hip/hip_runtime.h>
#include <hip/hip_bf16.h>
#include <math.h>

typedef __attribute__((ext_vector_type(16))) __bf16 v16bf;
typedef __attribute__((ext_vector_type(8)))  __bf16 v8bf;
typedef __attribute__((ext_vector_type(4)))  __bf16 v4bf;
typedef __attribute__((ext_vector_type(8)))  float  v8f;
typedef __attribute__((ext_vector_type(4)))  int    v4i;

#define BDIM 512
#define TDIM 64
#define OBSD 1024
#define ACTD 32
#define HD   1024
#define SD   64

#define LDSTRIDE 40   // bf16 elements per LDS row (32 data + 8 pad); 80B, 16B-aligned

// CDNA5 async global->LDS path (ASYNCcnt-tracked, no VGPR round trip)
#if defined(__gfx1250__) && __has_builtin(__builtin_amdgcn_global_load_async_to_lds_b128) && \
    __has_builtin(__builtin_amdgcn_s_wait_asynccnt)
#define USE_ASYNC_LDS 1
#else
#define USE_ASYNC_LDS 0
#endif

// ---------------------------------------------------------------------------
// WMMA GEMM:  C[M,N] = act( concat(A0[:,0:K1], A1[:,...])[M,Ktot] * W^T + bias )
// A row-major f32 (two segments, boundaries 32-aligned); W row-major bf16 [N,Ktot].
// Block: 256 threads (8 waves). Tile 64(M) x 128(N); wave computes 16x64 (4 frags).
// Double-buffered LDS; W staged via GLOBAL_LOAD_ASYNC_TO_LDS_B128 when available.
// Intended for M%64==0, N%128==0 (edge lanes are clamped, results never stored OOB).
// ---------------------------------------------------------------------------
__global__ __launch_bounds__(256) void gemm_wmma_bf16(
    const float* __restrict__ A0, int lda0, int K1,
    const float* __restrict__ A1, int lda1,
    int Ktot,
    const __bf16* __restrict__ W,
    const float* __restrict__ bias,
    float* __restrict__ C, int ldc,
    int M, int N, int act)
{
  __shared__ __bf16 sA[2][64 * LDSTRIDE];
  __shared__ __bf16 sW[2][128 * LDSTRIDE];

  const int tid   = threadIdx.x;
  const int lane  = tid & 31;
  const int wave  = tid >> 5;
  const int wm    = wave & 3;
  const int wn    = wave >> 2;
  const int lhalf = (lane >= 16) ? 1 : 0;
  const int r16   = lane & 15;

  const int nBase   = blockIdx.x * 128;
  const int rowBase = blockIdx.y * 64;

  // staging split: A tile = 512 float4 (2/thread), W tile = 512 x 16B (2/thread)
  const int ar0 = tid >> 3;            // 0..31
  const int ar1 = ar0 + 32;            // 32..63
  const int ac  = (tid & 7) * 4;       // 0,4,..,28
  const int wr0 = tid >> 2;            // 0..63
  const int wr1 = wr0 + 64;            // 64..127
  const int wc  = (tid & 3) * 8;       // 0,8,16,24

  // clamped global row/col indices (memory-safe; clamped lanes never stored)
  const int ag0 = min(rowBase + ar0, M - 1);
  const int ag1 = min(rowBase + ar1, M - 1);
  const int wg0 = min(nBase + wr0, N - 1);
  const int wg1 = min(nBase + wr1, N - 1);

  v8f acc[4];
#pragma unroll
  for (int f = 0; f < 4; ++f)
#pragma unroll
    for (int i = 0; i < 8; ++i) acc[f][i] = 0.0f;

  const int nk = Ktot >> 5;

  float4 va0, va1;
#if !USE_ASYNC_LDS
  v8bf vw0, vw1;
#endif

  // ---- A: global f32 loads into registers for chunk kc ----
  auto gloadA = [&](int kc) {
    const int k0 = kc * 32;
    const float* Ap; int lda; int kloc;
    if (k0 < K1) { Ap = A0; lda = lda0; kloc = k0; }
    else         { Ap = A1; lda = lda1; kloc = k0 - K1; }
    va0 = *(const float4*)(Ap + (size_t)ag0 * lda + kloc + ac);
    va1 = *(const float4*)(Ap + (size_t)ag1 * lda + kloc + ac);
#if !USE_ASYNC_LDS
    vw0 = *(const v8bf*)(W + (size_t)wg0 * Ktot + k0 + wc);
    vw1 = *(const v8bf*)(W + (size_t)wg1 * Ktot + k0 + wc);
#endif
  };

  // ---- W: async global->LDS deposit for chunk kc into buffer `buf` ----
  auto stageW = [&](int kc, int buf) {
#if USE_ASYNC_LDS
    const int k0 = kc * 32;
    __builtin_amdgcn_global_load_async_to_lds_b128(
        (v4i*)(W + (size_t)wg0 * Ktot + k0 + wc),
        (v4i*)&sW[buf][wr0 * LDSTRIDE + wc], 0, 0);
    __builtin_amdgcn_global_load_async_to_lds_b128(
        (v4i*)(W + (size_t)wg1 * Ktot + k0 + wc),
        (v4i*)&sW[buf][wr1 * LDSTRIDE + wc], 0, 0);
#else
    (void)kc; (void)buf;
#endif
  };

  // ---- commit A (convert) and, in fallback mode, W registers into LDS ----
  auto lstore = [&](int buf) {
    v4bf h0, h1;
    h0[0] = (__bf16)va0.x; h0[1] = (__bf16)va0.y; h0[2] = (__bf16)va0.z; h0[3] = (__bf16)va0.w;
    h1[0] = (__bf16)va1.x; h1[1] = (__bf16)va1.y; h1[2] = (__bf16)va1.z; h1[3] = (__bf16)va1.w;
    *(v4bf*)&sA[buf][ar0 * LDSTRIDE + ac] = h0;
    *(v4bf*)&sA[buf][ar1 * LDSTRIDE + ac] = h1;
#if !USE_ASYNC_LDS
    *(v8bf*)&sW[buf][wr0 * LDSTRIDE + wc] = vw0;
    *(v8bf*)&sW[buf][wr1 * LDSTRIDE + wc] = vw1;
#endif
  };

  // prologue: chunk 0 -> buf 0
  if (nk > 0) { gloadA(0); stageW(0, 0); lstore(0); }

  int cur = 0;
  for (int kc = 0; kc < nk; ++kc) {
#if USE_ASYNC_LDS
    __builtin_amdgcn_s_wait_asynccnt(0);   // own async deposits done ...
#endif
    __syncthreads();                        // ... then visible to all waves
    const bool pre = (kc + 1 < nk);
    if (pre) {
      gloadA(kc + 1);                       // A loads in flight during compute
      stageW(kc + 1, cur ^ 1);              // W deposits directly into LDS
    }

    // A fragment: contiguous bf16 runs (ISA 7.12.2 16-bit A 16x32 layout)
    const __bf16* ap = &sA[cur][(wm * 16 + r16) * LDSTRIDE] + (lhalf ? 8 : 0);
    v8bf alo = *(const v8bf*)ap;
    v8bf ahi = *(const v8bf*)(ap + 16);
    v16bf a;
#pragma unroll
    for (int e = 0; e < 8; ++e) { a[e] = alo[e]; a[e + 8] = ahi[e]; }

    // issue ALL B fragment loads first, then the 4 independent WMMAs:
    // lets the scheduler use partial s_wait_dscnt and pipeline the matrix ops
    v16bf bfr[4];
#pragma unroll
    for (int f = 0; f < 4; ++f) {
      const __bf16* bp = &sW[cur][(wn * 64 + f * 16 + r16) * LDSTRIDE] + (lhalf ? 8 : 0);
      v8bf blo = *(const v8bf*)bp;
      v8bf bhi = *(const v8bf*)(bp + 16);
#pragma unroll
      for (int e = 0; e < 8; ++e) { bfr[f][e] = blo[e]; bfr[f][e + 8] = bhi[e]; }
    }
#pragma unroll
    for (int f = 0; f < 4; ++f) {
      acc[f] = __builtin_amdgcn_wmma_f32_16x16x32_bf16(
          false, a, false, bfr[f], (short)0, acc[f], false, false);
    }

    if (pre) lstore(cur ^ 1);               // A convert+commit behind the WMMAs
    cur ^= 1;
  }

  // epilogue: C layout = VGPR i -> M = i + 8*lhalf, N = lane%16
  const int rowInTile = wm * 16 + (lhalf ? 8 : 0);
#pragma unroll
  for (int f = 0; f < 4; ++f) {
    int gcol = nBase + wn * 64 + f * 16 + r16;
    if (gcol < N) {
      float bv = bias[gcol];
#pragma unroll
      for (int i = 0; i < 8; ++i) {
        int grow = rowBase + rowInTile + i;
        if (grow < M) {
          float v = acc[f][i] + bv;
          if (act == 1) v = fmaxf(v, 0.0f);
          C[(size_t)grow * ldc + gcol] = v;
        }
      }
    }
  }
}

// ---------------------------------------------------------------------------
// one-shot f32 -> bf16 weight conversion (n multiple of 4)
__global__ void cvt_f32_bf16(const float* __restrict__ src, __bf16* __restrict__ dst, int n) {
  int i = (blockIdx.x * blockDim.x + threadIdx.x) * 4;
  if (i + 3 < n) {
    float4 v = *(const float4*)(src + i);
    v4bf h;
    h[0] = (__bf16)v.x; h[1] = (__bf16)v.y; h[2] = (__bf16)v.z; h[3] = (__bf16)v.w;
    *(v4bf*)(dst + i) = h;
  }
}

__global__ void init_state(float* deter, float* stoch, float* klslot, int nD, int nS) {
  int idx = blockIdx.x * blockDim.x + threadIdx.x;
  if (idx < nD) deter[idx] = 0.0f;
  if (idx < nS) stoch[idx] = 0.0f;
  if (idx == 0) *klslot = 0.0f;
}

__device__ __forceinline__ float sigmoidf_(float x) { return 1.0f / (1.0f + expf(-x)); }

__global__ void gru_gate(const float* __restrict__ gi, const float* __restrict__ gh,
                         float* __restrict__ deter, int BH) {
  int idx = blockIdx.x * blockDim.x + threadIdx.x;
  if (idx >= BH) return;
  int b = idx / HD, j = idx - b * HD;
  const float* gib = gi + (size_t)b * 3 * HD;
  const float* ghb = gh + (size_t)b * 3 * HD;
  float r = sigmoidf_(gib[j]          + ghb[j]);
  float z = sigmoidf_(gib[HD + j]     + ghb[HD + j]);
  float n = tanhf    (gib[2 * HD + j] + r * ghb[2 * HD + j]);
  deter[idx] = (1.0f - z) * n + z * deter[idx];
}

__device__ __forceinline__ float softplusf_(float x) {
  return (x > 20.0f) ? x : log1pf(expf(x));
}

__global__ void stats_sample(const float* __restrict__ priorb, const float* __restrict__ postb,
                             const float* __restrict__ eps_t, int eps_stride,
                             float* __restrict__ stoch, float* __restrict__ klslot, float scale) {
  __shared__ float red[256];
  int idx = blockIdx.x * blockDim.x + threadIdx.x;
  float kl = 0.0f;
  if (idx < BDIM * SD) {
    int b = idx / SD, s = idx - b * SD;
    float pm = priorb[b * 2 * SD + s];
    float ps = softplusf_(priorb[b * 2 * SD + SD + s]) + 1e-4f;
    float qm = postb [b * 2 * SD + s];
    float qs = softplusf_(postb [b * 2 * SD + SD + s]) + 1e-4f;
    float e  = eps_t[(size_t)b * eps_stride + s];
    stoch[idx] = qm + e * qs;
    float dm = qm - pm;
    kl = logf(ps / qs) + (qs * qs + dm * dm) / (2.0f * ps * ps) - 0.5f;
  }
  red[threadIdx.x] = kl;
  __syncthreads();
  for (int s = 128; s > 0; s >>= 1) {
    if (threadIdx.x < s) red[threadIdx.x] += red[threadIdx.x + s];
    __syncthreads();
  }
  if (threadIdx.x == 0) atomicAdd(klslot, red[0] * scale);
}

__global__ __launch_bounds__(256) void heads_kernel(
    const float* __restrict__ deter, const float* __restrict__ stoch,
    const float* __restrict__ Wr, const float* __restrict__ br,
    const float* __restrict__ Wd, const float* __restrict__ bd,
    float* __restrict__ rew_out, float* __restrict__ done_out, int t) {
  int wave = threadIdx.x >> 5;
  int lane = threadIdx.x & 31;
  int b = blockIdx.x * 8 + wave;
  if (b >= BDIM) return;
  float sr = 0.0f, sd = 0.0f;
  const float* db = deter + (size_t)b * HD;
  for (int k = lane; k < HD; k += 32) { float a = db[k]; sr += a * Wr[k]; sd += a * Wd[k]; }
  const float* sb = stoch + (size_t)b * SD;
  for (int k = lane; k < SD; k += 32) { float a = sb[k]; sr += a * Wr[HD + k]; sd += a * Wd[HD + k]; }
  for (int off = 16; off > 0; off >>= 1) { sr += __shfl_xor(sr, off); sd += __shfl_xor(sd, off); }
  if (lane == 0) {
    rew_out [(size_t)b * TDIM + t] = sr + br[0];
    done_out[(size_t)b * TDIM + t] = sd + bd[0];
  }
}

// ---------------------------------------------------------------------------
static inline void launch_gemm(const float* A0, int lda0, int K1,
                               const float* A1, int lda1, int Ktot,
                               const __bf16* W, const float* bias,
                               float* C, int ldc, int M, int N, int act,
                               hipStream_t s) {
  dim3 grid((N + 127) / 128, (M + 63) / 64);
  gemm_wmma_bf16<<<grid, 256, 0, s>>>(A0, lda0, K1, A1, lda1, Ktot, W, bias, C, ldc, M, N, act);
}

static inline void launch_cvt(const float* src, __bf16* dst, size_t n, hipStream_t s) {
  cvt_f32_bf16<<<(unsigned)((n / 4 + 255) / 256), 256, 0, s>>>(src, dst, (int)n);
}

extern "C" void kernel_launch(void* const* d_in, const int* in_sizes, int n_in,
                              void* d_out, int out_size, void* d_ws, size_t ws_size,
                              hipStream_t stream) {
  (void)in_sizes; (void)n_in; (void)out_size;
  const float* obs     = (const float*)d_in[0];
  const float* action  = (const float*)d_in[1];
  const float* eps     = (const float*)d_in[2];
  const float* W_enc   = (const float*)d_in[3];
  const float* b_enc   = (const float*)d_in[4];
  const float* W_ih    = (const float*)d_in[5];
  const float* W_hh    = (const float*)d_in[6];
  const float* b_ih    = (const float*)d_in[7];
  const float* b_hh    = (const float*)d_in[8];
  const float* W_prior = (const float*)d_in[9];
  const float* b_prior = (const float*)d_in[10];
  const float* W_post  = (const float*)d_in[11];
  const float* b_post  = (const float*)d_in[12];
  const float* W_dec1  = (const float*)d_in[13];
  const float* b_dec1  = (const float*)d_in[14];
  const float* W_dec2  = (const float*)d_in[15];
  const float* b_dec2  = (const float*)d_in[16];
  const float* W_rew   = (const float*)d_in[17];
  const float* b_rew   = (const float*)d_in[18];
  const float* W_done  = (const float*)d_in[19];
  const float* b_done  = (const float*)d_in[20];

  float* pred   = (float*)d_out;                     // [B,T,OBS]
  float* reward = pred + (size_t)BDIM * TDIM * OBSD; // [B,T]
  float* doneo  = reward + (size_t)BDIM * TDIM;      // [B,T]
  float* klslot = doneo + (size_t)BDIM * TDIM;       // [1]

  // ---- workspace: f32 activations, then bf16 weights, then (optional) emb_all
  float* w = (float*)d_ws;
  size_t off = 0;
  float* deter  = w + off; off += (size_t)BDIM * HD;
  float* stoch  = w + off; off += (size_t)BDIM * SD;
  float* gi     = w + off; off += (size_t)BDIM * 3 * HD;
  float* gh     = w + off; off += (size_t)BDIM * 3 * HD;
  float* priorb = w + off; off += (size_t)BDIM * 2 * SD;
  float* postb  = w + off; off += (size_t)BDIM * 2 * SD;
  float* hidden = w + off; off += (size_t)BDIM * HD;

  __bf16* hb = (__bf16*)(w + off);
  size_t hoff = 0;
  const size_t nEnc  = (size_t)HD * OBSD;
  const size_t nIh   = (size_t)3 * HD * (SD + ACTD);
  const size_t nHh   = (size_t)3 * HD * HD;
  const size_t nPri  = (size_t)2 * SD * HD;
  const size_t nPost = (size_t)2 * SD * 2 * HD;
  const size_t nD1   = (size_t)HD * (HD + SD);
  const size_t nD2   = (size_t)OBSD * HD;
  __bf16* Wenc_h  = hb + hoff; hoff += nEnc;
  __bf16* Wih_h   = hb + hoff; hoff += nIh;
  __bf16* Whh_h   = hb + hoff; hoff += nHh;
  __bf16* Wpri_h  = hb + hoff; hoff += nPri;
  __bf16* Wpost_h = hb + hoff; hoff += nPost;
  __bf16* Wd1_h   = hb + hoff; hoff += nD1;
  __bf16* Wd2_h   = hb + hoff; hoff += nD2;

  float* emb = (float*)(hb + hoff);
  const size_t used_bytes = off * sizeof(float) + hoff * sizeof(__bf16);
  const bool full_emb =
      ws_size >= used_bytes + (size_t)BDIM * TDIM * HD * sizeof(float);

  // ---- per-call init: recurrent state, kl accumulator, bf16 weights
  {
    int n = BDIM * HD;
    init_state<<<(n + 255) / 256, 256, 0, stream>>>(deter, stoch, klslot, n, BDIM * SD);
  }
  launch_cvt(W_enc,   Wenc_h,  nEnc,  stream);
  launch_cvt(W_ih,    Wih_h,   nIh,   stream);
  launch_cvt(W_hh,    Whh_h,   nHh,   stream);
  launch_cvt(W_prior, Wpri_h,  nPri,  stream);
  launch_cvt(W_post,  Wpost_h, nPost, stream);
  launch_cvt(W_dec1,  Wd1_h,   nD1,   stream);
  launch_cvt(W_dec2,  Wd2_h,   nD2,   stream);

  // obs encoder up-front if workspace allows (one big parallel GEMM)
  if (full_emb) {
    launch_gemm(obs, OBSD, BDIM * TDIM * OBSD, nullptr, 0, OBSD,
                Wenc_h, b_enc, emb, HD, BDIM * TDIM, HD, 1, stream);
  }

  const float kscale = 1.0f / ((float)BDIM * (float)TDIM * (float)SD);

  for (int t = 0; t < TDIM; ++t) {
    const float* emb_t;
    int emb_ld;
    if (full_emb) { emb_t = emb + (size_t)t * HD; emb_ld = TDIM * HD; }
    else {
      launch_gemm(obs + (size_t)t * OBSD, TDIM * OBSD, OBSD, nullptr, 0, OBSD,
                  Wenc_h, b_enc, emb, HD, BDIM, HD, 1, stream);
      emb_t = emb; emb_ld = HD;
    }

    // gi = [stoch, a_t] @ W_ih^T + b_ih
    launch_gemm(stoch, SD, SD, action + (size_t)t * ACTD, TDIM * ACTD, SD + ACTD,
                Wih_h, b_ih, gi, 3 * HD, BDIM, 3 * HD, 0, stream);
    // gh = deter @ W_hh^T + b_hh
    launch_gemm(deter, HD, HD, nullptr, 0, HD,
                Whh_h, b_hh, gh, 3 * HD, BDIM, 3 * HD, 0, stream);
    gru_gate<<<(BDIM * HD + 255) / 256, 256, 0, stream>>>(gi, gh, deter, BDIM * HD);

    // prior / posterior stats
    launch_gemm(deter, HD, HD, nullptr, 0, HD,
                Wpri_h, b_prior, priorb, 2 * SD, BDIM, 2 * SD, 0, stream);
    launch_gemm(deter, HD, HD, emb_t, emb_ld, 2 * HD,
                Wpost_h, b_post, postb, 2 * SD, BDIM, 2 * SD, 0, stream);

    stats_sample<<<(BDIM * SD + 255) / 256, 256, 0, stream>>>(
        priorb, postb, eps + (size_t)t * SD, TDIM * SD, stoch, klslot, kscale);

    // decoder
    launch_gemm(deter, HD, HD, stoch, SD, HD + SD,
                Wd1_h, b_dec1, hidden, HD, BDIM, HD, 1, stream);
    launch_gemm(hidden, HD, HD, nullptr, 0, HD,
                Wd2_h, b_dec2, pred + (size_t)t * OBSD, TDIM * OBSD, BDIM, OBSD, 0, stream);

    heads_kernel<<<BDIM / 8, 256, 0, stream>>>(deter, stoch, W_rew, b_rew,
                                               W_done, b_done, reward, doneo, t);
  }
}
